// SageLayer_89979564851208
// MI455X (gfx1250) — compile-verified, matrix operand order checked by reading, not run on hardware
//
#include <hip/hip_runtime.h>
#include <hip/hip_bf16.h>

typedef __attribute__((ext_vector_type(16))) __bf16 v16bf;
typedef __attribute__((ext_vector_type(8)))  __bf16 v8bf;
typedef __attribute__((ext_vector_type(8)))  float  v8f;
typedef unsigned int u32x4 __attribute__((ext_vector_type(4)));
typedef int          i32x4 __attribute__((ext_vector_type(4)));
typedef int          i32x8 __attribute__((ext_vector_type(8)));

#define BB   16
#define NTOK 1024
#define DD   1024
#define HH   4096
#define EE   8
#define TOPK 2

#if defined(__HIP_DEVICE_COMPILE__) && __has_builtin(__builtin_amdgcn_tensor_load_to_lds)
#define USE_TDM 1
#else
#define USE_TDM 0
#endif

static __device__ __forceinline__ v16bf cat16(v8bf lo, v8bf hi){
  v16bf r;
#pragma unroll
  for (int i=0;i<8;i++){ r[i]=lo[i]; r[i+8]=hi[i]; }
  return r;
}

// branch-free approximate-gelu (tanh form), saturates correctly at +/-inf
static __device__ __forceinline__ float gelu_tanh(float x){
  float u = 0.7978845608028654f*(x + 0.044715f*x*x*x);
  float e = __expf(2.0f*u);                       // v_exp_f32
  float t = 1.0f - 2.0f*__builtin_amdgcn_rcpf(e + 1.0f);
  return 0.5f*x*(1.0f + t);
}

#if USE_TDM
// TDM 2D tile load: tile_cols contiguous bf16 elems per row, tile_rows rows,
// row stride = row_stride elems. Dest: contiguous [tile_rows][tile_cols] at lds_byte_off.
static __device__ __forceinline__ void tdm_load_tile(unsigned lds_byte_off,
                                                     const void* gptr,
                                                     unsigned tile_cols,
                                                     unsigned tile_rows,
                                                     unsigned row_stride)
{
  unsigned long long ga = (unsigned long long)gptr;
  u32x4 g0;
  g0[0] = 1u;                                    // count=1, user descriptor
  g0[1] = lds_byte_off;                          // lds_addr
  g0[2] = (unsigned)ga;                          // global_addr[31:0]
  g0[3] = (unsigned)((ga >> 32) & 0x01FFFFFFull) | (2u << 30);  // addr[56:32] | type=2
  const unsigned td0 = 0x40000000u, td1 = 0x40000000u;  // huge tensor dims: no OOB clip
  i32x8 g1;
  g1[0] = (int)(1u << 16);                       // wg_mask=0, data_size=1 (2B), no pad
  g1[1] = (int)((td0 & 0xFFFFu) << 16);          // [31:16]=tensor_dim0[15:0]
  g1[2] = (int)((td0 >> 16) | ((td1 & 0xFFFFu) << 16));  // td0[31:16] | td1[15:0]
  g1[3] = (int)((td1 >> 16) | (tile_cols << 16));        // td1[31:16] | tile_dim0
  g1[4] = (int)(tile_rows & 0xFFFFu);            // tile_dim1 | tile_dim2=0
  g1[5] = (int)row_stride;                       // tensor_dim0_stride[31:0] (elems)
  g1[6] = 0;                                     // stride0[47:32] | stride1[15:0]
  g1[7] = 0;                                     // stride1[47:16]
  i32x4 gz = {0,0,0,0};
#if __clang_major__ >= 23
  i32x8 gz8 = {0,0,0,0,0,0,0,0};
  __builtin_amdgcn_tensor_load_to_lds(g0, g1, gz, gz, gz8, 0);
#else
  __builtin_amdgcn_tensor_load_to_lds(g0, g1, gz, gz, 0);
#endif
}
#endif

// Double-buffered GEMM core.
// Block tile: M = MI*32, N = 128, K-step 32. 256 threads = 8 waves (2 in M x 4 in N).
// Wave tile: (MI*16) x 32 -> acc[MI][2] of v8f. A:[MI*32,K] rows, WT:[128,K] rows.
// LDS: lA = 2 buffers of (MI*32)*32 bf16, lB = 2 buffers of 128*32 bf16.
template<int MI>
static __device__ __forceinline__ void gemm_core(const __bf16* __restrict__ A,
                                                 const __bf16* __restrict__ WT,
                                                 int Kdim, v8f acc[MI][2],
                                                 __bf16* lA, __bf16* lB)
{
  constexpr int AR   = MI*32;        // A rows in block tile
  constexpr int ALD  = AR*32;        // A elems per LDS buffer
  constexpr int BLD  = 128*32;       // B elems per LDS buffer
  const int tid  = threadIdx.x;
  const int lane = tid & 31;
  const int wave = tid >> 5;
  const int wm   = (wave & 1) * (MI*16);
  const int wn   = (wave >> 1) * 32;
  const int lrow = lane & 15;
  const int kh   = (lane >> 4) * 8;  // per-lane K sub-offset (0 or 8)

#if USE_TDM
  const unsigned baseA = (unsigned)(unsigned long long)(const void*)lA;
  const unsigned baseB = (unsigned)(unsigned long long)(const void*)lB;
  if (wave == 0){
    tdm_load_tile(baseA, A,  32, AR,  (unsigned)Kdim);
    tdm_load_tile(baseB, WT, 32, 128, (unsigned)Kdim);
  }
  int cur = 0;
  for (int k0 = 0; k0 < Kdim; k0 += 32){
    if (wave == 0) __builtin_amdgcn_s_wait_tensorcnt((short)0);
    __syncthreads();                      // publish buffer `cur` to all waves
    const bool more = (k0 + 32) < Kdim;
    if (more && wave == 0){               // prefetch next tile into other buffer via TDM
      tdm_load_tile(baseA + (unsigned)((cur^1)*ALD*2), A  + k0 + 32, 32, AR,  (unsigned)Kdim);
      tdm_load_tile(baseB + (unsigned)((cur^1)*BLD*2), WT + k0 + 32, 32, 128, (unsigned)Kdim);
    }
    const __bf16* cA = lA + cur*ALD;
    const __bf16* cB = lB + cur*BLD;
    v16bf af[MI], bfr[2];
#pragma unroll
    for (int i = 0; i < MI; i++){
      const __bf16* p = cA + (wm + i*16 + lrow)*32 + kh;  // K chunks {kh..kh+7, kh+16..kh+23}
      af[i] = cat16(*(const v8bf*)p, *(const v8bf*)(p + 16));
    }
#pragma unroll
    for (int j = 0; j < 2; j++){
      const __bf16* q = cB + (wn + j*16 + lrow)*32 + kh;
      bfr[j] = cat16(*(const v8bf*)q, *(const v8bf*)(q + 16));
    }
#pragma unroll
    for (int i = 0; i < MI; i++)
#pragma unroll
      for (int j = 0; j < 2; j++)
        acc[i][j] = __builtin_amdgcn_wmma_f32_16x16x32_bf16(
            false, af[i], false, bfr[j], (short)0, acc[i][j], false, false);
    if (more) cur ^= 1;
  }
#else
  // fallback: per-thread register staging
  constexpr int ACH = AR*4;
  v8bf ra[ACH/256], rb[2];
#pragma unroll
  for (int t = 0; t < ACH/256; t++){
    int id = tid + t*256, r = id >> 2, c = (id & 3) << 3;
    ra[t] = *(const v8bf*)(A + (size_t)r*Kdim + c);
  }
#pragma unroll
  for (int t = 0; t < 2; t++){
    int id = tid + t*256, r = id >> 2, c = (id & 3) << 3;
    rb[t] = *(const v8bf*)(WT + (size_t)r*Kdim + c);
  }
#pragma unroll
  for (int t = 0; t < ACH/256; t++){
    int id = tid + t*256, r = id >> 2, c = (id & 3) << 3;
    *(v8bf*)(lA + r*32 + c) = ra[t];
  }
#pragma unroll
  for (int t = 0; t < 2; t++){
    int id = tid + t*256, r = id >> 2, c = (id & 3) << 3;
    *(v8bf*)(lB + r*32 + c) = rb[t];
  }
  int cur = 0;
  for (int k0 = 0; k0 < Kdim; k0 += 32){
    __syncthreads();
    const bool more = (k0 + 32) < Kdim;
    if (more){
      const __bf16* An = A  + k0 + 32;
      const __bf16* Wn = WT + k0 + 32;
#pragma unroll
      for (int t = 0; t < ACH/256; t++){
        int id = tid + t*256, r = id >> 2, c = (id & 3) << 3;
        ra[t] = *(const v8bf*)(An + (size_t)r*Kdim + c);
      }
#pragma unroll
      for (int t = 0; t < 2; t++){
        int id = tid + t*256, r = id >> 2, c = (id & 3) << 3;
        rb[t] = *(const v8bf*)(Wn + (size_t)r*Kdim + c);
      }
    }
    const __bf16* cA = lA + cur*ALD;
    const __bf16* cB = lB + cur*BLD;
    v16bf af[MI], bfr[2];
#pragma unroll
    for (int i = 0; i < MI; i++){
      const __bf16* p = cA + (wm + i*16 + lrow)*32 + kh;
      af[i] = cat16(*(const v8bf*)p, *(const v8bf*)(p + 16));
    }
#pragma unroll
    for (int j = 0; j < 2; j++){
      const __bf16* q = cB + (wn + j*16 + lrow)*32 + kh;
      bfr[j] = cat16(*(const v8bf*)q, *(const v8bf*)(q + 16));
    }
#pragma unroll
    for (int i = 0; i < MI; i++)
#pragma unroll
      for (int j = 0; j < 2; j++)
        acc[i][j] = __builtin_amdgcn_wmma_f32_16x16x32_bf16(
            false, af[i], false, bfr[j], (short)0, acc[i][j], false, false);
    if (more){
      __bf16* nA = lA + (cur^1)*ALD;
      __bf16* nB = lB + (cur^1)*BLD;
#pragma unroll
      for (int t = 0; t < ACH/256; t++){
        int id = tid + t*256, r = id >> 2, c = (id & 3) << 3;
        *(v8bf*)(nA + r*32 + c) = ra[t];
      }
#pragma unroll
      for (int t = 0; t < 2; t++){
        int id = tid + t*256, r = id >> 2, c = (id & 3) << 3;
        *(v8bf*)(nB + r*32 + c) = rb[t];
      }
      cur ^= 1;
    }
  }
#endif
}

// ---------- conversion kernels ----------
__global__ void k_cvt_bf16(const float* __restrict__ in, __bf16* __restrict__ out, int n){
  int i = blockIdx.x*256 + threadIdx.x;
  if (i < n) out[i] = (__bf16)in[i];
}

// src f32 [Krows, Ncols] (+z*K*N) -> dst bf16 [Ncols, Krows] (+z*K*N)
__global__ void k_tcvt(const float* __restrict__ src, __bf16* __restrict__ dst,
                       int Krows, int Ncols){
  __shared__ float tile[32][33];
  size_t zo = (size_t)blockIdx.z * Krows * Ncols;
  src += zo; dst += zo;
  int n0 = blockIdx.x*32, k0 = blockIdx.y*32;
  int tx = threadIdx.x & 31, ty = threadIdx.x >> 5;
  for (int r = ty; r < 32; r += 8)
    tile[r][tx] = src[(size_t)(k0+r)*Ncols + n0 + tx];
  __syncthreads();
  for (int r = ty; r < 32; r += 8)
    dst[(size_t)(n0+r)*Krows + k0 + tx] = (__bf16)tile[tx][r];
}

// ---------- router ----------
__global__ void k_meanpool(const float* __restrict__ x, float* __restrict__ meanb){
  int b = blockIdx.y;
  int d = blockIdx.x*256 + threadIdx.x;
  const float* xb = x + (size_t)b*NTOK*DD + d;
  float s = 0.f;
  for (int n = 0; n < NTOK; n++) s += xb[(size_t)n*DD];
  meanb[b*DD + d] = s * (1.0f/NTOK);
}

__global__ void k_router(const float* __restrict__ meanb, const float* __restrict__ w_gate,
                         const float* __restrict__ b_gate, int* __restrict__ idx,
                         float* __restrict__ gw){
  int b = blockIdx.x;
  int lane = threadIdx.x & 31, e = threadIdx.x >> 5;   // 8 waves, one expert each
  const float* m = meanb + b*DD;
  float p = 0.f;
  for (int d = lane; d < DD; d += 32) p += m[d] * w_gate[(size_t)d*EE + e];
  for (int off = 16; off; off >>= 1) p += __shfl_down(p, off, 32);
  __shared__ float logits[EE];
  if (lane == 0) logits[e] = p + b_gate[e];
  __syncthreads();
  if (threadIdx.x == 0){
    float lmax = -1e30f;
    for (int i = 0; i < EE; i++) lmax = fmaxf(lmax, logits[i]);
    float pr[EE], s = 0.f;
    for (int i = 0; i < EE; i++){ pr[i] = __expf(logits[i]-lmax); s += pr[i]; }
    int i0 = 0;
    for (int i = 1; i < EE; i++) if (pr[i] > pr[i0]) i0 = i;
    int i1 = (i0 == 0) ? 1 : 0;
    for (int i = 0; i < EE; i++) if (i != i0 && pr[i] > pr[i1]) i1 = i;
    float v0 = pr[i0]/s, v1 = pr[i1]/s, t = v0 + v1;
    idx[b*TOPK+0] = i0; idx[b*TOPK+1] = i1;
    gw[b*TOPK+0] = v0/t; gw[b*TOPK+1] = v1/t;
  }
}

// ---------- GEMM kernels (MI=4: block tile 128x128) ----------
__global__ __launch_bounds__(256) void k_gemm1_main(const __bf16* __restrict__ xb,
    const __bf16* __restrict__ W1T, const float* __restrict__ b1,
    __bf16* __restrict__ hout){
  __shared__ __bf16 lA[2*128*32];
  __shared__ __bf16 lB[2*128*32];
  int bx = blockIdx.x, by = blockIdx.y;
  const __bf16* A  = xb  + (size_t)by*128*DD;
  const __bf16* WT = W1T + (size_t)bx*128*DD;
  v8f acc[4][2] = {};
  gemm_core<4>(A, WT, DD, acc, lA, lB);
  int lane = threadIdx.x & 31, wave = threadIdx.x >> 5;
  int wm=(wave&1)*64, wn=(wave>>1)*32, nc=lane&15, mh=(lane>>4)*8;
#pragma unroll
  for (int i=0;i<4;i++)
#pragma unroll
  for (int j=0;j<2;j++){
    int n  = bx*128 + wn + j*16 + nc;
    float bv = b1[n];
    int mb = by*128 + wm + i*16 + mh;
#pragma unroll
    for (int r=0;r<8;r++)
      hout[(size_t)(mb+r)*HH + n] = (__bf16)gelu_tanh(acc[i][j][r] + bv);
  }
}

__global__ __launch_bounds__(256) void k_gemm2_main(const __bf16* __restrict__ hm,
    const __bf16* __restrict__ W2T, const float* __restrict__ b2,
    const float* __restrict__ alpha_p, float* __restrict__ out){
  __shared__ __bf16 lA[2*128*32];
  __shared__ __bf16 lB[2*128*32];
  int bx = blockIdx.x, by = blockIdx.y;
  float a = fminf(fmaxf(alpha_p[0], 0.1f), 1.0f);
  const __bf16* A  = hm  + (size_t)by*128*HH;
  const __bf16* WT = W2T + (size_t)bx*128*HH;
  v8f acc[4][2] = {};
  gemm_core<4>(A, WT, HH, acc, lA, lB);
  int lane = threadIdx.x & 31, wave = threadIdx.x >> 5;
  int wm=(wave&1)*64, wn=(wave>>1)*32, nc=lane&15, mh=(lane>>4)*8;
#pragma unroll
  for (int i=0;i<4;i++)
#pragma unroll
  for (int j=0;j<2;j++){
    int n  = bx*128 + wn + j*16 + nc;
    float bv = b2[n];
    int mb = by*128 + wm + i*16 + mh;
#pragma unroll
    for (int r=0;r<8;r++)
      out[(size_t)(mb+r)*DD + n] = a * (acc[i][j][r] + bv);
  }
}

__global__ __launch_bounds__(256) void k_gemm1_exp(const __bf16* __restrict__ xb,
    const __bf16* __restrict__ W1eT, const float* __restrict__ b1e,
    const int* __restrict__ idx, __bf16* __restrict__ he){
  __shared__ __bf16 lA[2*128*32];
  __shared__ __bf16 lB[2*128*32];
  int bx = blockIdx.x, by = blockIdx.y, z = blockIdx.z;
  int b = z >> 1, e = idx[z];
  const __bf16* A  = xb   + ((size_t)b*NTOK + by*128)*DD;
  const __bf16* WT = W1eT + (size_t)e*DD*HH + (size_t)bx*128*DD;
  const float* bias = b1e + (size_t)e*HH;
  __bf16* outz = he + (size_t)z*NTOK*HH;
  v8f acc[4][2] = {};
  gemm_core<4>(A, WT, DD, acc, lA, lB);
  int lane = threadIdx.x & 31, wave = threadIdx.x >> 5;
  int wm=(wave&1)*64, wn=(wave>>1)*32, nc=lane&15, mh=(lane>>4)*8;
#pragma unroll
  for (int i=0;i<4;i++)
#pragma unroll
  for (int j=0;j<2;j++){
    int n  = bx*128 + wn + j*16 + nc;
    float bv = bias[n];
    int mb = by*128 + wm + i*16 + mh;
#pragma unroll
    for (int r=0;r<8;r++)
      outz[(size_t)(mb+r)*HH + n] = (__bf16)gelu_tanh(acc[i][j][r] + bv);
  }
}

// out[b,n,d] += (1-a) * sum_k gw[b,k] * (he[b,k] @ W2e[idx]^T + b2e[idx])
// MI=2 (block tile 64x128) to hold two accumulator sets without spilling.
__global__ __launch_bounds__(256) void k_exp2_comb(const __bf16* __restrict__ he,
    const __bf16* __restrict__ W2eT, const float* __restrict__ b2e,
    const int* __restrict__ idx, const float* __restrict__ gw,
    const float* __restrict__ alpha_p, float* __restrict__ out){
  __shared__ __bf16 lA[2*64*32];
  __shared__ __bf16 lB[2*128*32];
  int bx = blockIdx.x, by = blockIdx.y, b = blockIdx.z;
  float a = fminf(fmaxf(alpha_p[0], 0.1f), 1.0f);
  float oma = 1.0f - a;
  int lane = threadIdx.x & 31, wave = threadIdx.x >> 5;
  int wm=(wave&1)*32, wn=(wave>>1)*32, nc=lane&15, mh=(lane>>4)*8;
  v8f tot[2][2] = {};
#pragma unroll
  for (int k = 0; k < TOPK; k++){
    int z = b*TOPK + k, e = idx[z];
    float g = gw[z] * oma;
    const __bf16* A  = he   + ((size_t)z*NTOK + by*64)*HH;
    const __bf16* WT = W2eT + (size_t)e*DD*HH + (size_t)bx*128*HH;
    const float* bias = b2e + (size_t)e*DD;
    v8f acc[2][2] = {};
    gemm_core<2>(A, WT, HH, acc, lA, lB);
#pragma unroll
    for (int i=0;i<2;i++)
#pragma unroll
    for (int j=0;j<2;j++){
      float bv = bias[bx*128 + wn + j*16 + nc];
#pragma unroll
      for (int r=0;r<8;r++)
        tot[i][j][r] += g * (acc[i][j][r] + bv);
    }
    __syncthreads();   // protect LDS reuse across k iterations
  }
#pragma unroll
  for (int i=0;i<2;i++)
#pragma unroll
  for (int j=0;j<2;j++){
    int n  = bx*128 + wn + j*16 + nc;
    int mb = by*64 + wm + i*16 + mh;
#pragma unroll
    for (int r=0;r<8;r++){
      size_t o = ((size_t)b*NTOK + mb + r)*DD + n;
      out[o] += tot[i][j][r];
    }
  }
}

extern "C" void kernel_launch(void* const* d_in, const int* in_sizes, int n_in,
                              void* d_out, int out_size, void* d_ws, size_t ws_size,
                              hipStream_t stream) {
  (void)in_sizes; (void)n_in; (void)out_size; (void)ws_size;
  const float* x      = (const float*)d_in[0];
  const float* alpha  = (const float*)d_in[1];
  const float* w_gate = (const float*)d_in[2];
  const float* b_gate = (const float*)d_in[3];
  const float* W1m    = (const float*)d_in[4];
  const float* b1m    = (const float*)d_in[5];
  const float* W2m    = (const float*)d_in[6];
  const float* b2m    = (const float*)d_in[7];
  const float* W1e    = (const float*)d_in[8];
  const float* b1e    = (const float*)d_in[9];
  const float* W2e    = (const float*)d_in[10];
  const float* b2e    = (const float*)d_in[11];
  float* out = (float*)d_out;

  // workspace carve-up (256B aligned)
  char* ws = (char*)d_ws;
  size_t off = 0;
  auto carve = [&](size_t bytes)->char*{
    char* p = ws + off; off = (off + bytes + 255) & ~(size_t)255; return p;
  };
  __bf16* xb    = (__bf16*)carve((size_t)BB*NTOK*DD*2);        // 32 MB
  __bf16* W1mT  = (__bf16*)carve((size_t)DD*HH*2);             // 8 MB
  __bf16* W2mT  = (__bf16*)carve((size_t)DD*HH*2);             // 8 MB
  __bf16* W1eT  = (__bf16*)carve((size_t)EE*DD*HH*2);          // 64 MB
  __bf16* W2eT  = (__bf16*)carve((size_t)EE*DD*HH*2);          // 64 MB
  __bf16* hm    = (__bf16*)carve((size_t)BB*NTOK*HH*2);        // 128 MB
  __bf16* he    = (__bf16*)carve((size_t)BB*TOPK*NTOK*HH*2);   // 256 MB
  float*  meanb = (float*) carve((size_t)BB*DD*4);
  int*    idxb  = (int*)   carve((size_t)BB*TOPK*4);
  float*  gwb   = (float*) carve((size_t)BB*TOPK*4);

  // 1) precision conversion + weight transposes
  int nx = BB*NTOK*DD;
  k_cvt_bf16<<<(nx+255)/256, 256, 0, stream>>>(x, xb, nx);
  k_tcvt<<<dim3(HH/32, DD/32, 1),  256, 0, stream>>>(W1m, W1mT, DD, HH);
  k_tcvt<<<dim3(DD/32, HH/32, 1),  256, 0, stream>>>(W2m, W2mT, HH, DD);
  k_tcvt<<<dim3(HH/32, DD/32, EE), 256, 0, stream>>>(W1e, W1eT, DD, HH);
  k_tcvt<<<dim3(DD/32, HH/32, EE), 256, 0, stream>>>(W2e, W2eT, HH, DD);

  // 2) router
  k_meanpool<<<dim3(DD/256, BB), 256, 0, stream>>>(x, meanb);
  k_router<<<BB, 256, 0, stream>>>(meanb, w_gate, b_gate, idxb, gwb);

  // 3) main path
  k_gemm1_main<<<dim3(HH/128, (BB*NTOK)/128), 256, 0, stream>>>(xb, W1mT, b1m, hm);
  k_gemm2_main<<<dim3(DD/128, (BB*NTOK)/128), 256, 0, stream>>>(hm, W2mT, b2m, alpha, out);

  // 4) expert path
  k_gemm1_exp<<<dim3(HH/128, NTOK/128, BB*TOPK), 256, 0, stream>>>(xb, W1eT, b1e, idxb, he);
  k_exp2_comb<<<dim3(DD/128, NTOK/64, BB), 256, 0, stream>>>(he, W2eT, b2e, idxb, gwb, alpha, out);
}
